// ConceptAttentionLayer_74775380623815
// MI455X (gfx1250) — compile-verified, matrix operand order checked by reading, not run on hardware
//
#include <hip/hip_runtime.h>
#include <hip/hip_bf16.h>
#include <math.h>

// ---------------------------------------------------------------------------
// Types for CDNA5 WMMA (wave32, v_wmma_f32_16x16x32_bf16)
// ---------------------------------------------------------------------------
typedef __attribute__((ext_vector_type(16))) __bf16 v16bf;
typedef __attribute__((ext_vector_type(8)))  __bf16 v8bf;
typedef __attribute__((ext_vector_type(8)))  float  v8f;

#define HIDC  128
#define NHEAD 8
#define NNODE 65536
#define NCONC 8192
#define NEDGE 393216

__device__ inline float  bf2f(__bf16 x) { return (float)x; }
__device__ inline __bf16 f2bf(float x)  { return (__bf16)x; }

__device__ inline v8f wmma_bf16(v16bf a, v16bf b, v8f c) {
  // D = A(16x32 bf16) x B(32x16 bf16) + C(16x16 f32)
  return __builtin_amdgcn_wmma_f32_16x16x32_bf16(
      false, a, false, b, (short)0, c, false, false);
}

__device__ inline v16bf combine16(v8bf lo, v8bf hi) {
  v16bf r;
#pragma unroll
  for (int j = 0; j < 8; ++j) { r[j] = lo[j]; r[j + 8] = hi[j]; }
  return r;
}

// A fragment: 16x32 (MxK) bf16. Lane holds row M = lane&15; K halves split by
// lane>>4 per the CDNA5 16-bit A layout (VGPR0-3: K=0..7 / 8..15, VGPR4-7:
// K=16..23 / 24..31). rowbase already points at this lane's row.
__device__ inline v16bf load_a_frag(const __bf16* rowbase, int k0, int hh) {
  const __bf16* p = rowbase + k0 + hh * 8;
  v8bf lo = *(const v8bf*)p;
  v8bf hi = *(const v8bf*)(p + 16);
  return combine16(lo, hi);
}

// B fragment: 32x16 (KxN) with B[k][n] = W[n][k], W row-major [Mout][K].
__device__ inline v16bf load_b_frag(const __bf16* W, int ldw, int n0, int k0,
                                    int lane) {
  const __bf16* p = W + (long)(n0 + (lane & 15)) * ldw + k0 + (lane >> 4) * 8;
  v8bf lo = *(const v8bf*)p;
  v8bf hi = *(const v8bf*)(p + 16);
  return combine16(lo, hi);
}

__device__ inline void atomicMaxF(float* a, float v) {
  unsigned int* p = (unsigned int*)a;
  unsigned int cur = *p;
  while (__uint_as_float(cur) < v) {
    unsigned int assumed = cur;
    cur = atomicCAS(p, assumed, __float_as_uint(v));
    if (cur == assumed) break;
  }
}

// ---------------------------------------------------------------------------
// Elementwise utility kernels
// ---------------------------------------------------------------------------
__global__ void f32_to_bf16_kernel(const float* __restrict__ s,
                                   __bf16* __restrict__ d, long n) {
  long gid = (long)blockIdx.x * 256 + threadIdx.x;
  if (gid < n) d[gid] = f2bf(s[gid]);
}

__global__ void fill_kernel(float* __restrict__ p, float v, long n) {
  long gid = (long)blockIdx.x * 256 + threadIdx.x;
  if (gid < n) p[gid] = v;
}

__global__ void tile_concepts_kernel(const float* __restrict__ cq,
                                     float* __restrict__ out) {
  long gid = (long)blockIdx.x * 256 + threadIdx.x;  // NCONC*128 threads
  out[gid] = cq[gid & 1023];                        // tile 8x128 queries
}

__global__ void cat_kernel(const float* __restrict__ agg,
                           const __bf16* __restrict__ xd,
                           __bf16* __restrict__ cat) {
  long gid = (long)blockIdx.x * 256 + threadIdx.x;  // N*128
  long n = gid >> 7;
  int  c = (int)(gid & 127);
  cat[n * 256 + c]       = f2bf(agg[gid]);
  cat[n * 256 + 128 + c] = xd[gid];
}

__global__ void gate_upd_kernel(const float* __restrict__ glin,
                                const float* __restrict__ agg,
                                const float* __restrict__ s,
                                __bf16* __restrict__ upd) {
  long gid = (long)blockIdx.x * 256 + threadIdx.x;
  float g = 1.f / (1.f + __expf(-glin[gid]));
  upd[gid] = f2bf(agg[gid] + g * (s[gid] - agg[gid]));
}

__global__ void final_mask_kernel(const unsigned char* __restrict__ mask,
                                  const float* __restrict__ feat,
                                  const float* __restrict__ xn,
                                  float* __restrict__ out) {
  long gid = (long)blockIdx.x * 256 + threadIdx.x;
  long n = gid >> 7;
  out[gid] = mask[n] ? feat[gid] : xn[gid];
}

// ---------------------------------------------------------------------------
// LayerNorm over rows of 128; optional residual add (out = res + LN(x)),
// optional f32 and/or bf16 outputs. One wave per row, 8 rows per block.
// ---------------------------------------------------------------------------
__global__ __launch_bounds__(256) void ln_kernel(
    const float* __restrict__ x, const float* __restrict__ g,
    const float* __restrict__ b, const float* __restrict__ res,
    float* __restrict__ outf, __bf16* __restrict__ outb) {
  int row  = blockIdx.x * 8 + (threadIdx.x >> 5);
  int lane = threadIdx.x & 31;
  const float* xr = x + (long)row * HIDC;
  float4 xv = *(const float4*)(xr + lane * 4);
  float s = xv.x + xv.y + xv.z + xv.w;
#pragma unroll
  for (int o = 16; o > 0; o >>= 1) s += __shfl_xor(s, o, 32);
  float mean = s * (1.f / HIDC);
  float dx = xv.x - mean, dy = xv.y - mean, dz = xv.z - mean, dw = xv.w - mean;
  float ss = dx * dx + dy * dy + dz * dz + dw * dw;
#pragma unroll
  for (int o = 16; o > 0; o >>= 1) ss += __shfl_xor(ss, o, 32);
  float rstd = rsqrtf(ss * (1.f / HIDC) + 1e-5f);
  int c = lane * 4;
  long base = (long)row * HIDC + c;
  float y0 = dx * rstd * g[c + 0] + b[c + 0];
  float y1 = dy * rstd * g[c + 1] + b[c + 1];
  float y2 = dz * rstd * g[c + 2] + b[c + 2];
  float y3 = dw * rstd * g[c + 3] + b[c + 3];
  if (res) {
    y0 += res[base + 0]; y1 += res[base + 1];
    y2 += res[base + 2]; y3 += res[base + 3];
  }
  if (outf) {
    outf[base + 0] = y0; outf[base + 1] = y1;
    outf[base + 2] = y2; outf[base + 3] = y3;
  }
  if (outb) {
    outb[base + 0] = f2bf(y0); outb[base + 1] = f2bf(y1);
    outb[base + 2] = f2bf(y2); outb[base + 3] = f2bf(y3);
  }
}

// ---------------------------------------------------------------------------
// Generic WMMA GEMM: C[N][Mout] = act(A_bf16[N][K] @ W_bf16[Mout][K]^T + bias)
// One wave -> 16 rows x 128 cols. flags: 1=relu, 2=accumulate into Cf.
// N multiple of 128, Mout multiple of 128 (grid.y chunks), K multiple of 32.
// All 8 B fragments of a k-tile are loaded into registers before the WMMA
// burst so the loads clause together and the 8 independent WMMAs issue
// back-to-back instead of each draining LOADcnt.
// ---------------------------------------------------------------------------
__global__ __launch_bounds__(256) void gemm_kernel(
    const __bf16* __restrict__ A, int K, const __bf16* __restrict__ W,
    const float* __restrict__ bias, float* __restrict__ Cf,
    __bf16* __restrict__ Cb, int Mout, int flags) {
  int wave = threadIdx.x >> 5, lane = threadIdx.x & 31;
  int row0 = (blockIdx.x * 8 + wave) * 16;
  int c0   = blockIdx.y * 128;
  int m    = lane & 15, hh = lane >> 4;
  const __bf16* arow = A + (long)(row0 + m) * K;

  v8f zero;
#pragma unroll
  for (int i = 0; i < 8; ++i) zero[i] = 0.f;
  v8f acc[8];
#pragma unroll
  for (int n = 0; n < 8; ++n) acc[n] = zero;

  int nk = K >> 5;
  for (int kt = 0; kt < nk; ++kt) {
    v16bf af = load_a_frag(arow, kt * 32, hh);
    v16bf bfr[8];
#pragma unroll
    for (int n = 0; n < 8; ++n)
      bfr[n] = load_b_frag(W, K, c0 + n * 16, kt * 32, lane);
#pragma unroll
    for (int n = 0; n < 8; ++n) acc[n] = wmma_bf16(af, bfr[n], acc[n]);
  }
#pragma unroll
  for (int n = 0; n < 8; ++n) {
    int col = c0 + n * 16 + m;
    float bb = bias ? bias[col] : 0.f;
#pragma unroll
    for (int i = 0; i < 8; ++i) {
      long idx = (long)(row0 + i + 8 * hh) * Mout + col;
      float val = acc[n][i] + bb;
      if (flags & 1) val = fmaxf(val, 0.f);
      if (flags & 2) val += Cf[idx];
      if (Cf) Cf[idx] = val;
      if (Cb) Cb[idx] = f2bf(val);
    }
  }
}

// ---------------------------------------------------------------------------
// Stage-1 fused edge kernel. One wave handles 16 edges:
//   rn = LN(r[e])                         (single pass over 201MB r_s1)
//   kr = rn @ Wkr^T, vr = rn @ Wvr^T      (WMMA, 64 wmma per tile; the two
//                                          accumulator chains interleave)
//   k_j = k[src]+kr; sim[h] = 0.25*<q[dst],k_j> (butterfly reduce)
//   atomicMax(mx); v_j = v[src]+vr -> bf16 ws
// ---------------------------------------------------------------------------
__global__ __launch_bounds__(256) void edge_s1_kernel(
    const float* __restrict__ r, const float* __restrict__ lng,
    const float* __restrict__ lnb, const __bf16* __restrict__ Wkr,
    const __bf16* __restrict__ Wvr, const int* __restrict__ src,
    const int* __restrict__ dst, const float* __restrict__ q,
    const float* __restrict__ kmat, const float* __restrict__ vmat,
    float* __restrict__ sim, float* __restrict__ mx, __bf16* __restrict__ vj) {
  __shared__ __bf16 atile[8][16][HIDC];
  __shared__ int sidx[8][16];
  __shared__ int didx[8][16];

  const int wave = threadIdx.x >> 5;
  const int lane = threadIdx.x & 31;
  const int lrow = lane & 15;   // edge row within tile / N index
  const int hh   = lane >> 4;   // half
  const long e0  = ((long)blockIdx.x * 8 + wave) * 16;

  if (hh == 0) {
    sidx[wave][lrow] = src[e0 + lrow];
    didx[wave][lrow] = dst[e0 + lrow];
  }

  // --- LayerNorm of r rows (lane covers 64 contiguous floats of its row) ---
  const float* rr = r + (e0 + lrow) * HIDC + hh * 64;
  float s1 = 0.f, s2 = 0.f;
#pragma unroll
  for (int jj = 0; jj < 16; ++jj) {
    float4 t = *(const float4*)(rr + jj * 4);
    s1 += t.x + t.y + t.z + t.w;
    s2 += t.x * t.x + t.y * t.y + t.z * t.z + t.w * t.w;
  }
  s1 += __shfl_xor(s1, 16, 32);
  s2 += __shfl_xor(s2, 16, 32);
  float mean = s1 * (1.f / HIDC);
  float var  = s2 * (1.f / HIDC) - mean * mean;
  float rstd = rsqrtf(fmaxf(var, 0.f) + 1e-5f);
#pragma unroll
  for (int jj = 0; jj < 16; ++jj) {
    float4 t = *(const float4*)(rr + jj * 4);
    int c = hh * 64 + jj * 4;
    atile[wave][lrow][c + 0] = f2bf((t.x - mean) * rstd * lng[c + 0] + lnb[c + 0]);
    atile[wave][lrow][c + 1] = f2bf((t.y - mean) * rstd * lng[c + 1] + lnb[c + 1]);
    atile[wave][lrow][c + 2] = f2bf((t.z - mean) * rstd * lng[c + 2] + lnb[c + 2]);
    atile[wave][lrow][c + 3] = f2bf((t.w - mean) * rstd * lng[c + 3] + lnb[c + 3]);
  }
  __syncthreads();

  // --- A fragments (16x128 tile -> 4 k-tiles) ---
  v16bf afr[4];
  const __bf16* abase = &atile[wave][lrow][0];
#pragma unroll
  for (int kt = 0; kt < 4; ++kt) afr[kt] = load_a_frag(abase, kt * 32, hh);

  v8f zero;
#pragma unroll
  for (int i = 0; i < 8; ++i) zero[i] = 0.f;

  const int col16 = lrow;  // channel-within-head index for this lane
#pragma unroll 1
  for (int n = 0; n < NHEAD; ++n) {
    int col = n * 16 + col16;
    // Batch-load all 8 weight fragments (4 kr + 4 vr), then run the two
    // independent WMMA accumulator chains interleaved for ILP.
    v16bf bk[4], bv[4];
#pragma unroll
    for (int kt = 0; kt < 4; ++kt) {
      bk[kt] = load_b_frag(Wkr, HIDC, n * 16, kt * 32, lane);
      bv[kt] = load_b_frag(Wvr, HIDC, n * 16, kt * 32, lane);
    }
    v8f kacc = zero, vacc = zero;
#pragma unroll
    for (int kt = 0; kt < 4; ++kt) {
      kacc = wmma_bf16(afr[kt], bk[kt], kacc);
      vacc = wmma_bf16(afr[kt], bv[kt], vacc);
    }
    // k_j = kr + k[src], sim = 0.25 * <q[dst], k_j>
#pragma unroll
    for (int i = 0; i < 8; ++i) {
      int  m  = i + 8 * hh;
      int  sI = sidx[wave][m];
      int  dI = didx[wave][m];
      float kj = kacc[i] + kmat[(long)sI * HIDC + col];
      float p  = q[(long)dI * HIDC + col] * kj;
      p += __shfl_xor(p, 1, 32);
      p += __shfl_xor(p, 2, 32);
      p += __shfl_xor(p, 4, 32);
      p += __shfl_xor(p, 8, 32);
      if (col16 == 0) {
        float sv = p * 0.25f;  // HD^-0.5
        sim[(e0 + m) * NHEAD + n] = sv;
        atomicMaxF(&mx[(long)dI * NHEAD + n], sv);
      }
    }
    // v_j = vr + v[src] -> bf16 workspace
#pragma unroll
    for (int i = 0; i < 8; ++i) {
      int m  = i + 8 * hh;
      int sI = sidx[wave][m];
      vj[(e0 + m) * HIDC + col] =
          f2bf(vacc[i] + vmat[(long)sI * HIDC + col]);
    }
  }
}

// Stage-2 sim: one thread per (edge, head), 16-wide dot of gathered q/k.
__global__ __launch_bounds__(256) void edge_sim_kernel(
    const int* __restrict__ src, const int* __restrict__ dst,
    const float* __restrict__ q, const float* __restrict__ k,
    float* __restrict__ sim, float* __restrict__ mx) {
  long gid = (long)blockIdx.x * 256 + threadIdx.x;  // E*8 threads
  long e = gid >> 3;
  int  h = (int)(gid & 7);
  long si = src[e], di = dst[e];
  const float4* qp = (const float4*)(q + di * HIDC + h * 16);
  const float4* kp = (const float4*)(k + si * HIDC + h * 16);
  float acc = 0.f;
#pragma unroll
  for (int t = 0; t < 4; ++t) {
    float4 a = qp[t], b = kp[t];
    acc += a.x * b.x + a.y * b.y + a.z * b.z + a.w * b.w;
  }
  acc *= 0.25f;
  sim[gid] = acc;
  atomicMaxF(&mx[di * NHEAD + h], acc);
}

// Softmax pass 2: ex = exp(sim - mx[dst]); den += ex (segment sum).
__global__ __launch_bounds__(256) void softmax_ex_kernel(
    float* __restrict__ sim, const float* __restrict__ mx,
    float* __restrict__ den, const int* __restrict__ dst) {
  long gid = (long)blockIdx.x * 256 + threadIdx.x;  // E*8
  long e = gid >> 3;
  int  h = (int)(gid & 7);
  long d = dst[e];
  float m = mx[d * NHEAD + h];
  if (!__builtin_isfinite(m)) m = 0.f;  // matches reference isfinite guard
  float ex = __expf(sim[gid] - m);
  sim[gid] = ex;
  atomicAdd(&den[d * NHEAD + h], ex);
}

// Pass 3: scatter-add attn-weighted values into agg[dst].
// One thread per (edge, 4 channels). vj (bf16, per-edge) or vmat[src].
__global__ __launch_bounds__(256) void msg_kernel(
    const float* __restrict__ ex, const float* __restrict__ den,
    const int* __restrict__ src, const int* __restrict__ dst,
    const __bf16* __restrict__ vj, const float* __restrict__ vmat,
    float* __restrict__ agg) {
  long gid = (long)blockIdx.x * 256 + threadIdx.x;  // E*32
  long e  = gid >> 5;
  int  c0 = (int)(gid & 31) * 4;
  int  h  = c0 >> 4;
  long d  = dst[e];
  float a = ex[e * NHEAD + h] / (den[d * NHEAD + h] + 1e-16f);
  if (vj) {
    const __bf16* vp = vj + e * HIDC + c0;
#pragma unroll
    for (int t = 0; t < 4; ++t)
      atomicAdd(&agg[d * HIDC + c0 + t], a * bf2f(vp[t]));
  } else {
    const float* vp = vmat + (long)src[e] * HIDC + c0;
#pragma unroll
    for (int t = 0; t < 4; ++t)
      atomicAdd(&agg[d * HIDC + c0 + t], a * vp[t]);
  }
}

// ---------------------------------------------------------------------------
// Host-side orchestration
// ---------------------------------------------------------------------------
extern "C" void kernel_launch(void* const* d_in, const int* in_sizes, int n_in,
                              void* d_out, int out_size, void* d_ws,
                              size_t ws_size, hipStream_t stream) {
  (void)in_sizes; (void)n_in; (void)out_size; (void)ws_size;
  const float* feat_a = (const float*)d_in[0];
  const int*   ei1    = (const int*)d_in[1];  // [2, E]: src, dst
  const int*   ei2    = (const int*)d_in[2];
  const float* r_s1   = (const float*)d_in[3];
  const float* pos_q  = (const float*)d_in[4];
  const unsigned char* nmask = (const unsigned char*)d_in[5];
  const float* cq     = (const float*)d_in[7];
  const void* const* P1 = &d_in[8];   // params_s1: 28 tensors
  const void* const* P2 = &d_in[36];  // params_s2: 27 tensors
  auto PF1 = [&](int i) { return (const float*)P1[i]; };
  auto PF2 = [&](int i) { return (const float*)P2[i]; };

  // bump allocator on workspace
  char* wsb = (char*)d_ws;
  size_t off = 0;
  auto alloc = [&](size_t bytes) -> void* {
    void* p = wsb + off;
    off = (off + bytes + 255) & ~(size_t)255;
    return p;
  };
  auto conv = [&](const float* s, __bf16* d, long n) {
    f32_to_bf16_kernel<<<dim3((unsigned)((n + 255) / 256)), 256, 0, stream>>>(s, d, n);
  };
  auto fill = [&](float* p, float v, long n) {
    fill_kernel<<<dim3((unsigned)((n + 255) / 256)), 256, 0, stream>>>(p, v, n);
  };
  auto gemm = [&](const __bf16* A, int K, const __bf16* W, const float* bias,
                  float* Cf, __bf16* Cb, int N, int Mout, int flags) {
    gemm_kernel<<<dim3(N / 128, Mout / 128), 256, 0, stream>>>(A, K, W, bias,
                                                               Cf, Cb, Mout, flags);
  };
  auto ln = [&](const float* x, const float* g, const float* b,
                const float* res, float* of, __bf16* ob, int nrows) {
    ln_kernel<<<dim3(nrows / 8), 256, 0, stream>>>(x, g, b, res, of, ob);
  };

  // bf16 weight copies
  __bf16* wq1  = (__bf16*)alloc(16384 * 2); conv(PF1(0),  wq1,  16384);
  __bf16* wk1  = (__bf16*)alloc(16384 * 2); conv(PF1(2),  wk1,  16384);
  __bf16* wv1  = (__bf16*)alloc(16384 * 2); conv(PF1(3),  wv1,  16384);
  __bf16* ws1  = (__bf16*)alloc(16384 * 2); conv(PF1(4),  ws1,  16384);
  __bf16* wg1  = (__bf16*)alloc(32768 * 2); conv(PF1(6),  wg1,  32768);
  __bf16* wo1  = (__bf16*)alloc(16384 * 2); conv(PF1(8),  wo1,  16384);
  __bf16* w11  = (__bf16*)alloc(65536 * 2); conv(PF1(10), w11,  65536);
  __bf16* w21  = (__bf16*)alloc(65536 * 2); conv(PF1(12), w21,  65536);
  __bf16* wkr1 = (__bf16*)alloc(16384 * 2); conv(PF1(24), wkr1, 16384);
  __bf16* wvr1 = (__bf16*)alloc(16384 * 2); conv(PF1(25), wvr1, 16384);
  __bf16* wq2  = (__bf16*)alloc(16384 * 2); conv(PF2(0),  wq2,  16384);
  __bf16* wk2  = (__bf16*)alloc(16384 * 2); conv(PF2(2),  wk2,  16384);
  __bf16* wv2  = (__bf16*)alloc(16384 * 2); conv(PF2(3),  wv2,  16384);
  __bf16* ws2  = (__bf16*)alloc(16384 * 2); conv(PF2(4),  ws2,  16384);
  __bf16* wg2  = (__bf16*)alloc(32768 * 2); conv(PF2(6),  wg2,  32768);
  __bf16* wo2  = (__bf16*)alloc(16384 * 2); conv(PF2(8),  wo2,  16384);
  __bf16* w12  = (__bf16*)alloc(65536 * 2); conv(PF2(10), w12,  65536);
  __bf16* w22  = (__bf16*)alloc(65536 * 2); conv(PF2(12), w22,  65536);
  __bf16* wqr2 = (__bf16*)alloc(16384 * 2); conv(PF2(24), wqr2, 16384);

  // activation buffers
  const long NC = (long)NCONC * HIDC, NN = (long)NNODE * HIDC;
  float*  conc_a = (float*)alloc(NC * 4);
  float*  conc_b = (float*)alloc(NC * 4);
  __bf16* xs_bf  = (__bf16*)alloc(NN * 2);
  __bf16* xd_bf  = (__bf16*)alloc(NN * 2);
  __bf16* rq_bf  = (__bf16*)alloc(NN * 2);
  __bf16* xf_bf  = (__bf16*)alloc(NN * 2);
  __bf16* updbf  = (__bf16*)alloc(NN * 2);
  __bf16* catb   = (__bf16*)alloc((long)NNODE * 256 * 2);
  __bf16* h1     = (__bf16*)alloc((long)NNODE * 512 * 2);
  __bf16* vj1    = (__bf16*)alloc((long)NEDGE * HIDC * 2);
  float*  qbuf   = (float*)alloc(NN * 4);
  float*  kbuf   = (float*)alloc(NN * 4);
  float*  vbuf   = (float*)alloc(NN * 4);
  float*  sbuf   = (float*)alloc(NN * 4);
  float*  aggbuf = (float*)alloc(NN * 4);
  float*  tmp1   = (float*)alloc(NN * 4);
  float*  nodeA  = (float*)alloc(NN * 4);
  float*  nodeB  = (float*)alloc(NN * 4);
  float*  simb   = (float*)alloc((long)NEDGE * NHEAD * 4);
  float*  mxb    = (float*)alloc((long)NNODE * NHEAD * 4);
  float*  denb   = (float*)alloc((long)NNODE * NHEAD * 4);

  // ======================= Stage 1: units -> concepts ======================
  tile_concepts_kernel<<<dim3(NCONC * HIDC / 256), 256, 0, stream>>>(cq, conc_a);
  ln(feat_a, PF1(14), PF1(15), nullptr, nullptr, xs_bf, NNODE);   // ln_src
  ln(conc_a, PF1(16), PF1(17), nullptr, nullptr, xd_bf, NCONC);   // ln_dst
  gemm(xd_bf, 128, wq1, PF1(1), qbuf, nullptr, NCONC, 128, 0);    // q
  gemm(xs_bf, 128, wk1, nullptr, kbuf, nullptr, NNODE, 128, 0);   // k
  gemm(xs_bf, 128, wv1, nullptr, vbuf, nullptr, NNODE, 128, 0);   // v
  gemm(xd_bf, 128, ws1, PF1(5), sbuf, nullptr, NCONC, 128, 0);    // s
  fill(mxb, -INFINITY, (long)NCONC * NHEAD);
  fill(denb, 0.f, (long)NCONC * NHEAD);
  fill(aggbuf, 0.f, NC);
  edge_s1_kernel<<<dim3(NEDGE / 128), 256, 0, stream>>>(
      r_s1, PF1(26), PF1(27), wkr1, wvr1, ei1, ei1 + NEDGE, qbuf, kbuf, vbuf,
      simb, mxb, vj1);
  softmax_ex_kernel<<<dim3(NEDGE * NHEAD / 256), 256, 0, stream>>>(
      simb, mxb, denb, ei1 + NEDGE);
  msg_kernel<<<dim3(NEDGE * 32 / 256), 256, 0, stream>>>(
      simb, denb, ei1, ei1 + NEDGE, vj1, nullptr, aggbuf);
  cat_kernel<<<dim3((unsigned)(NC / 256)), 256, 0, stream>>>(aggbuf, xd_bf, catb);
  gemm(catb, 256, wg1, PF1(7), tmp1, nullptr, NCONC, 128, 0);     // gate lin
  gate_upd_kernel<<<dim3((unsigned)(NC / 256)), 256, 0, stream>>>(
      tmp1, aggbuf, sbuf, updbf);
  gemm(updbf, 128, wo1, PF1(9), tmp1, nullptr, NCONC, 128, 0);    // out proj
  ln(tmp1, PF1(18), PF1(19), conc_a, conc_b, nullptr, NCONC);     // x = xd+LN
  ln(conc_b, PF1(20), PF1(21), nullptr, nullptr, xf_bf, NCONC);   // ffpre
  gemm(xf_bf, 128, w11, PF1(11), nullptr, h1, NCONC, 512, 1);     // ff1+relu
  gemm(h1, 512, w21, PF1(13), tmp1, nullptr, NCONC, 128, 0);      // ff2
  ln(tmp1, PF1(22), PF1(23), conc_b, conc_a, nullptr, NCONC);     // x += LN

  // ======================= Stage 2: concepts -> units ======================
  ln(conc_a, PF2(14), PF2(15), nullptr, nullptr, xs_bf, NCONC);   // ln_src
  ln(feat_a, PF2(16), PF2(17), nullptr, nullptr, xd_bf, NNODE);   // ln_dst
  ln(pos_q, PF2(25), PF2(26), nullptr, nullptr, rq_bf, NNODE);    // ln_rq
  gemm(xd_bf, 128, wq2, PF2(1), qbuf, nullptr, NNODE, 128, 0);    // q
  gemm(rq_bf, 128, wqr2, nullptr, qbuf, nullptr, NNODE, 128, 2);  // q += rq@Wqr
  gemm(xs_bf, 128, wk2, nullptr, kbuf, nullptr, NCONC, 128, 0);   // k
  gemm(xs_bf, 128, wv2, nullptr, vbuf, nullptr, NCONC, 128, 0);   // v
  gemm(xd_bf, 128, ws2, PF2(5), sbuf, nullptr, NNODE, 128, 0);    // s
  fill(mxb, -INFINITY, (long)NNODE * NHEAD);
  fill(denb, 0.f, (long)NNODE * NHEAD);
  fill(aggbuf, 0.f, NN);
  edge_sim_kernel<<<dim3(NEDGE * NHEAD / 256), 256, 0, stream>>>(
      ei2, ei2 + NEDGE, qbuf, kbuf, simb, mxb);
  softmax_ex_kernel<<<dim3(NEDGE * NHEAD / 256), 256, 0, stream>>>(
      simb, mxb, denb, ei2 + NEDGE);
  msg_kernel<<<dim3(NEDGE * 32 / 256), 256, 0, stream>>>(
      simb, denb, ei2, ei2 + NEDGE, nullptr, vbuf, aggbuf);
  cat_kernel<<<dim3((unsigned)(NN / 256)), 256, 0, stream>>>(aggbuf, xd_bf, catb);
  gemm(catb, 256, wg2, PF2(7), tmp1, nullptr, NNODE, 128, 0);
  gate_upd_kernel<<<dim3((unsigned)(NN / 256)), 256, 0, stream>>>(
      tmp1, aggbuf, sbuf, updbf);
  gemm(updbf, 128, wo2, PF2(9), tmp1, nullptr, NNODE, 128, 0);
  ln(tmp1, PF2(18), PF2(19), feat_a, nodeA, nullptr, NNODE);
  ln(nodeA, PF2(20), PF2(21), nullptr, nullptr, xf_bf, NNODE);
  gemm(xf_bf, 128, w12, PF2(11), nullptr, h1, NNODE, 512, 1);
  gemm(h1, 512, w22, PF2(13), tmp1, nullptr, NNODE, 128, 0);
  ln(tmp1, PF2(22), PF2(23), nodeA, nodeB, nullptr, NNODE);

  final_mask_kernel<<<dim3((unsigned)(NN / 256)), 256, 0, stream>>>(
      nmask, feat_a, nodeB, (float*)d_out);
}
